// GCN_Net_ASAP_72060961292407
// MI455X (gfx1250) — compile-verified, matrix operand order checked by reading, not run on hardware
//
#include <hip/hip_runtime.h>
#include <math.h>

// ---------------------------------------------------------------------------
// GCN + ASAPooling pipeline for MI455X (gfx1250, wave32).
// Dense GEMMs -> v_wmma_f32_16x16x32_f16, fed by global_load_b128. KSTEPS is
// a template parameter so the K loop is fully unrolled into SSA straight-line
// code: no loop-carried register rotation, loads freely scheduled across
// WMMAs (s_clause batches), minimal hazard NOPs.
// Sparse graph ops -> global float atomics. Dense per-graph A aliases the
// dead h/h1/xqmax/xq region of the workspace.
// ---------------------------------------------------------------------------

typedef __attribute__((ext_vector_type(16))) _Float16 v16h;
typedef __attribute__((ext_vector_type(8)))  float    v8f;

constexpr int  BG  = 256;                 // graphs
constexpr int  NPG = 256;                 // nodes per graph
constexpr int  DEG = 8;
constexpr int  HID = 64;
constexpr int  KP  = 128;                 // kept nodes per graph
constexpr long NN  = (long)BG * NPG;      // 65536
constexpr long EE  = NN * DEG;            // 524288
constexpr long EF  = EE + NN;             // + self loops = 589824
constexpr long N64 = NN * HID;            // 4194304

// ---- workspace layout (float offsets; all multiples of 4 -> 16B aligned) ----
constexpr long OFF_H    = 0;
constexpr long OFF_H1   = OFF_H   + N64;
constexpr long OFF_XQM  = OFF_H1  + N64;
constexpr long OFF_XQ   = OFF_XQM + N64;
constexpr long OFF_ADEN = 0;                         // aliases H..XQ (4*N64 == B*NPG*NPG)
constexpr long OFF_XNEW = 4 * N64;
constexpr long OFF_DEG  = OFF_XNEW + N64;
constexpr long OFF_DINV = OFF_DEG  + NN;
constexpr long OFF_SMAX = OFF_DINV + NN;
constexpr long OFF_SSUM = OFF_SMAX + NN;
constexpr long OFF_AA   = OFF_SSUM + NN;
constexpr long OFF_BB   = OFF_AA   + NN;
constexpr long OFF_CC   = OFF_BB   + NN;
constexpr long OFF_FIT  = OFF_CC   + NN;
constexpr long OFF_S    = OFF_FIT  + NN;
constexpr long OFF_ESC  = OFF_S    + EF;
constexpr long OFF_SSLT = OFF_ESC  + EF;             // Ssel^T: [g][kp][srcLocal]
constexpr long OFF_AST  = OFF_SSLT + (long)BG * KP * NPG;  // AS^T: [g][kp][i]
constexpr long OFF_A2   = OFF_AST  + (long)BG * KP * NPG;  // A2:   [g][i][k'] row-major
constexpr long OFF_MT   = OFF_A2   + (long)BG * KP * KP;   // Mt:   [g][d][s] row-major
constexpr long OFF_XSEL = OFF_MT   + (long)BG * KP * KP;
constexpr long OFF_H2T  = OFF_XSEL + (long)BG * KP * HID;  // h2^T: [g][f][m]
constexpr long OFF_OUT2 = OFF_H2T  + (long)BG * KP * HID;  // out2: [g][d][f]
constexpr long OFF_D2   = OFF_OUT2 + (long)BG * KP * HID;
constexpr long OFF_W1T  = OFF_D2   + (long)BG * KP;        // 64x64 transposed weights
constexpr long OFF_LWT  = OFF_W1T  + 64 * 64;
constexpr long OFF_W2T  = OFF_LWT  + 64 * 64;
constexpr long OFF_INT  = OFF_W2T  + 64 * 64;        // ints: kpos[NN], perm[BG*KP]

// ---------------------------------------------------------------------------
// small helpers
// ---------------------------------------------------------------------------
__device__ __forceinline__ void atomicMaxF(float* addr, float val) {
  int* ai = (int*)addr;
  int old = __float_as_int(*addr);
  while (__int_as_float(old) < val) {
    int assumed = old;
    old = atomicCAS(ai, assumed, __float_as_int(val));
    if (old == assumed) break;
  }
}

__device__ __forceinline__ void edge_sd(long e, const int* src, const int* dst,
                                        int& s, int& d) {
  if (e < EE) { s = src[e]; d = dst[e]; }
  else        { int n = (int)(e - EE); s = n; d = n; }
}

__global__ void k_fill(float* __restrict__ p, float v, long n) {
  long i = (long)blockIdx.x * blockDim.x + threadIdx.x;
  if (i < n) p[i] = v;
}

__global__ void k_t64(const float* __restrict__ W, float* __restrict__ Wt) {
  int t = blockIdx.x * blockDim.x + threadIdx.x;       // 4096
  if (t >= 64 * 64) return;
  int n = t >> 6, k = t & 63;
  Wt[t] = W[k * 64 + n];
}

// ---------------------------------------------------------------------------
// WMMA GEMM, all operands K-contiguous (element stride 1).
// A element (m,k)  at A + m*a_sM + k   (16B-aligned rows)
// B element (k,n)  at B + n*b_sN + k   (stored N-major / K-contiguous)
// C element (m,n)  at C + m*c_sM + n*c_sN  (strides allow transposed store)
// One wave computes a 16x64 tile: 1 A-tile feeds 4 WMMAs per K-step; the
// whole K loop is fully unrolled (KSTEPS template), so the scheduler
// interleaves loads/converts/WMMAs without loop-carried copies.
// Per CDNA5 ISA 7.12.2: lane l<16 holds dim l, elements K = kb+{0..7,16..23},
// kb = 8*(lane>=16)  -> two contiguous float4 pairs per lane -> b128 loads.
// ---------------------------------------------------------------------------
__device__ __forceinline__ v16h ld_tile16(const float* __restrict__ base,
                                          long s_lane, int lane) {
  const int l  = lane & 15;
  const int kb = (lane >> 4) << 3;
  const float* p = base + (long)l * s_lane + kb;
  float4 q0 = *(const float4*)(p + 0);
  float4 q1 = *(const float4*)(p + 4);
  float4 q2 = *(const float4*)(p + 16);
  float4 q3 = *(const float4*)(p + 20);
  v16h r;
  r[0]  = (_Float16)q0.x; r[1]  = (_Float16)q0.y;
  r[2]  = (_Float16)q0.z; r[3]  = (_Float16)q0.w;
  r[4]  = (_Float16)q1.x; r[5]  = (_Float16)q1.y;
  r[6]  = (_Float16)q1.z; r[7]  = (_Float16)q1.w;
  r[8]  = (_Float16)q2.x; r[9]  = (_Float16)q2.y;
  r[10] = (_Float16)q2.z; r[11] = (_Float16)q2.w;
  r[12] = (_Float16)q3.x; r[13] = (_Float16)q3.y;
  r[14] = (_Float16)q3.z; r[15] = (_Float16)q3.w;
  return r;
}

template <int KSTEPS>
__global__ void k_wmma_gemm(const float* __restrict__ A, long a_sM, long a_bs,
                            const float* __restrict__ B, long b_sN, long b_bs,
                            const float* __restrict__ bias,
                            float* __restrict__ C, long c_sM, long c_sN, long c_bs,
                            int mtiles, int ngroups, int ntot) {
  int wave = threadIdx.x >> 5;
  int lane = threadIdx.x & 31;
  int tile = blockIdx.x * (blockDim.x >> 5) + wave;
  if (tile >= ntot) return;
  int per_batch = mtiles * ngroups;
  int b  = tile / per_batch;
  int r  = tile - b * per_batch;
  int mt = r / ngroups;
  int ng = r - mt * ngroups;

  const float* Ab = A + (long)b * a_bs + (long)(mt * 16) * a_sM;
  const float* Bb = B + (long)b * b_bs + (long)(ng * 64) * b_sN;

  v8f acc[4] = {v8f{}, v8f{}, v8f{}, v8f{}};
#pragma unroll
  for (int ks = 0; ks < KSTEPS; ++ks) {
    v16h av = ld_tile16(Ab + ks * 32, a_sM, lane);
#pragma unroll
    for (int j = 0; j < 4; ++j) {
      v16h bv = ld_tile16(Bb + (long)(j * 16) * b_sN + ks * 32, b_sN, lane);
      acc[j] = __builtin_amdgcn_wmma_f32_16x16x32_f16(false, av, false, bv,
                                                      (short)0, acc[j], false, false);
    }
  }

  int n  = lane & 15;
  int mh = (lane >> 4) << 3;
  float* Cb = C + (long)b * c_bs;
#pragma unroll
  for (int j = 0; j < 4; ++j) {
    int ncol = ng * 64 + j * 16 + n;
    float bb = bias ? bias[ncol] : 0.0f;
#pragma unroll
    for (int v = 0; v < 8; ++v)
      Cb[(long)(mt * 16 + mh + v) * c_sM + (long)ncol * c_sN] = acc[j][v] + bb;
  }
}

// ---------------------------------------------------------------------------
// conv1: degree, scatter, finalize
// ---------------------------------------------------------------------------
__global__ void k_count_deg(const int* __restrict__ dst, float* __restrict__ deg) {
  long e = (long)blockIdx.x * blockDim.x + threadIdx.x;
  if (e < EE) atomicAdd(&deg[dst[e]], 1.0f);
}
__global__ void k_deg_fin(float* __restrict__ deg, float* __restrict__ dinv) {
  long n = (long)blockIdx.x * blockDim.x + threadIdx.x;
  if (n >= NN) return;
  float d = deg[n] + 1.0f;   // self loop
  deg[n]  = d;               // == degF later
  dinv[n] = rsqrtf(d);
}
__global__ void k_conv1_scatter(const int* __restrict__ src, const int* __restrict__ dst,
                                const float* __restrict__ dinv,
                                const float* __restrict__ h, float* __restrict__ h1) {
  long t = (long)blockIdx.x * blockDim.x + threadIdx.x;
  if (t >= EE * HID) return;
  long e = t >> 6; int c = (int)(t & 63);
  int s = src[e], d = dst[e];
  float coef = dinv[s] * dinv[d];
  atomicAdd(&h1[(long)d * HID + c], h[(long)s * HID + c] * coef);
}
__global__ void k_conv1_fin(float* __restrict__ h1, const float* __restrict__ h,
                            const float* __restrict__ dinv, const float* __restrict__ b1) {
  long t = (long)blockIdx.x * blockDim.x + threadIdx.x;
  if (t >= N64) return;
  long n = t >> 6; int c = (int)(t & 63);
  float di = dinv[n];
  float v = h1[t] + h[t] * di * di + b1[c];
  h1[t] = v > 0.0f ? v : 0.0f;
}

// ---------------------------------------------------------------------------
// ASAP: segment max, attention, softmax, xnew
// ---------------------------------------------------------------------------
__global__ void k_segmax(const int* __restrict__ src, const int* __restrict__ dst,
                         const float* __restrict__ h1, float* __restrict__ xqm) {
  long t = (long)blockIdx.x * blockDim.x + threadIdx.x;
  if (t >= EF * HID) return;
  long e = t >> 6; int c = (int)(t & 63);
  int s, d; edge_sd(e, src, dst, s, d);
  atomicMaxF(&xqm[(long)d * HID + c], h1[(long)s * HID + c]);
}
__global__ void k_att(const int* __restrict__ src, const int* __restrict__ dst,
                      const float* __restrict__ xq, const float* __restrict__ h1,
                      const float* __restrict__ attW, const float* __restrict__ attb,
                      float* __restrict__ sOut) {
  long e = (long)blockIdx.x * blockDim.x + threadIdx.x;
  if (e >= EF) return;
  int s, d; edge_sd(e, src, dst, s, d);
  float acc = attb[0];
  const float* xr = xq + (long)d * HID;
  const float* hr = h1 + (long)s * HID;
#pragma unroll 4
  for (int c = 0; c < HID; ++c)
    acc += attW[c] * xr[c] + attW[HID + c] * hr[c];
  sOut[e] = acc > 0.0f ? acc : 0.2f * acc;   // leaky relu
}
__global__ void k_smax(const int* __restrict__ src, const int* __restrict__ dst,
                       const float* __restrict__ s, float* __restrict__ smax) {
  long e = (long)blockIdx.x * blockDim.x + threadIdx.x;
  if (e >= EF) return;
  int ss, d; edge_sd(e, src, dst, ss, d);
  atomicMaxF(&smax[d], s[e]);
}
__global__ void k_expsum(const int* __restrict__ src, const int* __restrict__ dst,
                         const float* __restrict__ s, const float* __restrict__ smax,
                         float* __restrict__ esc, float* __restrict__ ssum) {
  long e = (long)blockIdx.x * blockDim.x + threadIdx.x;
  if (e >= EF) return;
  int ss, d; edge_sd(e, src, dst, ss, d);
  float v = expf(s[e] - smax[d]);
  esc[e] = v;
  atomicAdd(&ssum[d], v);
}
__global__ void k_norm(const int* __restrict__ src, const int* __restrict__ dst,
                       float* __restrict__ esc, const float* __restrict__ ssum) {
  long e = (long)blockIdx.x * blockDim.x + threadIdx.x;
  if (e >= EF) return;
  int ss, d; edge_sd(e, src, dst, ss, d);
  esc[e] = esc[e] / (ssum[d] + 1e-16f);
}
__global__ void k_xnew(const int* __restrict__ src, const int* __restrict__ dst,
                       const float* __restrict__ h1, const float* __restrict__ score,
                       float* __restrict__ xnew) {
  long t = (long)blockIdx.x * blockDim.x + threadIdx.x;
  if (t >= EF * HID) return;
  long e = t >> 6; int c = (int)(t & 63);
  int s, d; edge_sd(e, src, dst, s, d);
  atomicAdd(&xnew[(long)d * HID + c], h1[(long)s * HID + c] * score[e]);
}

// ---------------------------------------------------------------------------
// LEConv fitness + top-k + x_sel
// ---------------------------------------------------------------------------
__global__ void k_ledots(const float* __restrict__ xnew,
                         const float* __restrict__ le1W, const float* __restrict__ le1b,
                         const float* __restrict__ le2W, const float* __restrict__ le3W,
                         const float* __restrict__ le3b,
                         float* __restrict__ aa, float* __restrict__ bb,
                         float* __restrict__ cc) {
  long n = (long)blockIdx.x * blockDim.x + threadIdx.x;
  if (n >= NN) return;
  const float* xr = xnew + n * HID;
  float a = le1b[0], b = 0.0f, c = le3b[0];
#pragma unroll 4
  for (int k = 0; k < HID; ++k) {
    float v = xr[k];
    a += v * le1W[k]; b += v * le2W[k]; c += v * le3W[k];
  }
  aa[n] = a; bb[n] = b; cc[n] = c;
}
__global__ void k_fitscatter(const int* __restrict__ src, const int* __restrict__ dst,
                             const float* __restrict__ aa, float* __restrict__ fit) {
  long e = (long)blockIdx.x * blockDim.x + threadIdx.x;
  if (e >= EF) return;
  int s, d; edge_sd(e, src, dst, s, d);
  atomicAdd(&fit[d], aa[s]);
}
__global__ void k_fitfin(float* __restrict__ fit, const float* __restrict__ degF,
                         const float* __restrict__ bb, const float* __restrict__ cc) {
  long n = (long)blockIdx.x * blockDim.x + threadIdx.x;
  if (n >= NN) return;
  float v = fit[n] - degF[n] * bb[n] + cc[n];
  fit[n] = 1.0f / (1.0f + expf(-v));
}
__global__ void k_topk(const float* __restrict__ fit, int* __restrict__ perm,
                       int* __restrict__ kpos) {
  __shared__ float f[NPG];
  int g = blockIdx.x, t = threadIdx.x;
  long node = (long)g * NPG + t;
  f[t] = fit[node];
  __syncthreads();
  float mine = f[t];
  int cnt = 0;
  for (int u = 0; u < NPG; ++u) {
    float fu = f[u];
    cnt += (fu > mine) || (fu == mine && u < t);
  }
  if (cnt < KP) { perm[g * KP + cnt] = (int)node; kpos[node] = cnt; }
  else          { kpos[node] = -1; }
}
__global__ void k_xsel(const int* __restrict__ perm, const float* __restrict__ xnew,
                       const float* __restrict__ fit, float* __restrict__ xsel) {
  long t = (long)blockIdx.x * blockDim.x + threadIdx.x;
  if (t >= (long)BG * KP * HID) return;
  long i = t >> 6; int c = (int)(t & 63);
  int node = perm[i];
  xsel[t] = xnew[(long)node * HID + c] * fit[node];
}

// ---------------------------------------------------------------------------
// coarsening scatters + conv2 helpers
// ---------------------------------------------------------------------------
__global__ void k_Ascatter(const int* __restrict__ src, const int* __restrict__ dst,
                           float* __restrict__ Aden) {
  long e = (long)blockIdx.x * blockDim.x + threadIdx.x;
  if (e >= EF) return;
  int s, d; edge_sd(e, src, dst, s, d);
  int g = d / NPG;
  int sL = s - g * NPG, dL = d - g * NPG;
  atomicAdd(&Aden[(long)g * NPG * NPG + (long)sL * NPG + dL], 1.0f);
}
// writes Ssel TRANSPOSED: SselT[g][kp][srcLocal]  (K-contiguous for WMMA)
__global__ void k_Sselscatter(const int* __restrict__ src, const int* __restrict__ dst,
                              const float* __restrict__ score, const int* __restrict__ kpos,
                              float* __restrict__ SselT) {
  long e = (long)blockIdx.x * blockDim.x + threadIdx.x;
  if (e >= EF) return;
  int s, d; edge_sd(e, src, dst, s, d);
  int kp = kpos[d];
  if (kp < 0) return;
  int g = d / NPG;
  int sL = s - g * NPG;
  atomicAdd(&SselT[((long)g * KP + kp) * NPG + sL], score[e]);
}
__global__ void k_d2(const float* __restrict__ A2, float* __restrict__ d2) {
  long t = (long)blockIdx.x * blockDim.x + threadIdx.x;
  if (t >= (long)BG * KP) return;
  int g = (int)(t / KP), d = (int)(t - (long)g * KP);
  const float* Ag = A2 + (long)g * KP * KP;
  float cnt = 0.0f;
  for (int s = 0; s < KP; ++s)
    cnt += (s == d) ? 1.0f : (Ag[(long)s * KP + d] > 0.0f ? 1.0f : 0.0f);
  d2[t] = rsqrtf(cnt);
}
__global__ void k_Mt(const float* __restrict__ A2, const float* __restrict__ d2,
                     float* __restrict__ Mt) {
  long t = (long)blockIdx.x * blockDim.x + threadIdx.x;
  if (t >= (long)BG * KP * KP) return;
  int g = (int)(t / (KP * KP));
  int r = (int)(t - (long)g * KP * KP);
  int dd = r / KP, ss = r - dd * KP;
  bool on = (ss == dd) || (A2[(long)g * KP * KP + (long)ss * KP + dd] > 0.0f);
  Mt[t] = on ? d2[(long)g * KP + ss] * d2[(long)g * KP + dd] : 0.0f;
}
__global__ void k_reduce(const float* __restrict__ out2, const float* __restrict__ b2,
                         float* __restrict__ out) {
  long t = (long)blockIdx.x * blockDim.x + threadIdx.x;
  if (t >= (long)BG * HID) return;
  int g = (int)(t >> 6), f = (int)(t & 63);
  const float* og = out2 + (long)g * KP * HID + f;
  float acc = 0.0f;
  for (int d = 0; d < KP; ++d) acc += og[(long)d * HID];
  out[t] = acc * (1.0f / KP) + b2[f];
}

// ---------------------------------------------------------------------------
// launch
// ---------------------------------------------------------------------------
extern "C" void kernel_launch(void* const* d_in, const int* in_sizes, int n_in,
                              void* d_out, int out_size, void* d_ws, size_t ws_size,
                              hipStream_t stream) {
  const float* x    = (const float*)d_in[0];
  const int*   ei   = (const int*)  d_in[1];
  const float* W1   = (const float*)d_in[3];
  const float* b1   = (const float*)d_in[4];
  const float* linW = (const float*)d_in[5];
  const float* linb = (const float*)d_in[6];
  const float* attW = (const float*)d_in[7];
  const float* attb = (const float*)d_in[8];
  const float* le1W = (const float*)d_in[9];
  const float* le1b = (const float*)d_in[10];
  const float* le2W = (const float*)d_in[11];
  const float* le3W = (const float*)d_in[12];
  const float* le3b = (const float*)d_in[13];
  const float* W2   = (const float*)d_in[14];
  const float* b2   = (const float*)d_in[15];
  float* out = (float*)d_out;
  float* w   = (float*)d_ws;

  const int* src = ei;
  const int* dst = ei + EE;

  float* h     = w + OFF_H;    float* h1   = w + OFF_H1;
  float* xqm   = w + OFF_XQM;  float* xq   = w + OFF_XQ;
  float* Aden  = w + OFF_ADEN; float* xnew = w + OFF_XNEW;
  float* deg   = w + OFF_DEG;  float* dinv = w + OFF_DINV;
  float* smax  = w + OFF_SMAX; float* ssum = w + OFF_SSUM;
  float* aa    = w + OFF_AA;   float* bb   = w + OFF_BB;
  float* cc    = w + OFF_CC;   float* fit  = w + OFF_FIT;
  float* sbuf  = w + OFF_S;    float* esc  = w + OFF_ESC;
  float* SselT = w + OFF_SSLT; float* ASt  = w + OFF_AST;
  float* A2    = w + OFF_A2;   float* Mt   = w + OFF_MT;
  float* xsel  = w + OFF_XSEL; float* h2t  = w + OFF_H2T;
  float* out2  = w + OFF_OUT2; float* d2   = w + OFF_D2;
  float* W1T   = w + OFF_W1T;  float* LWT  = w + OFF_LWT;
  float* W2T   = w + OFF_W2T;
  int*   kpos  = (int*)(w + OFF_INT);
  int*   perm  = kpos + NN;

  auto nb = [](long n, int t) { return dim3((unsigned)((n + t - 1) / t)); };

  auto gemm = [&](const float* A, long asM, long abs_,
                  const float* Bm, long bsN, long bbs,
                  const float* bias, float* C, long csM, long csN, long cbs,
                  int mt, int ngroups, int ks, int nbatch) {
    int tiles = nbatch * mt * ngroups;
    dim3 grid((unsigned)((tiles + 3) / 4)), blk(128);
    if (ks == 2)
      k_wmma_gemm<2><<<grid, blk, 0, stream>>>(A, asM, abs_, Bm, bsN, bbs, bias,
                                               C, csM, csN, cbs, mt, ngroups, tiles);
    else if (ks == 4)
      k_wmma_gemm<4><<<grid, blk, 0, stream>>>(A, asM, abs_, Bm, bsN, bbs, bias,
                                               C, csM, csN, cbs, mt, ngroups, tiles);
    else
      k_wmma_gemm<8><<<grid, blk, 0, stream>>>(A, asM, abs_, Bm, bsN, bbs, bias,
                                               C, csM, csN, cbs, mt, ngroups, tiles);
  };

  // ---- weight transposes (B operands must be N-major / K-contiguous) ----
  k_t64<<<nb(4096, 256), 256, 0, stream>>>(W1, W1T);
  k_t64<<<nb(4096, 256), 256, 0, stream>>>(linW, LWT);
  k_t64<<<nb(4096, 256), 256, 0, stream>>>(W2, W2T);

  // ---- conv1 ----
  k_fill<<<nb(NN, 256), 256, 0, stream>>>(deg, 0.0f, NN);
  k_count_deg<<<nb(EE, 256), 256, 0, stream>>>(dst, deg);
  k_deg_fin<<<nb(NN, 256), 256, 0, stream>>>(deg, dinv);

  // h = x @ W1 : M=NN, N=64 (1 ngroup), K=64 (2 ksteps)
  gemm(x, HID, 0, W1T, HID, 0, nullptr, h, HID, 1, 0, (int)(NN / 16), 1, 2, 1);

  k_fill<<<nb(N64, 256), 256, 0, stream>>>(h1, 0.0f, N64);
  k_conv1_scatter<<<nb(EE * HID, 256), 256, 0, stream>>>(src, dst, dinv, h, h1);
  k_conv1_fin<<<nb(N64, 256), 256, 0, stream>>>(h1, h, dinv, b1);

  // ---- ASAP attention ----
  k_fill<<<nb(N64, 256), 256, 0, stream>>>(xqm, -3.0e38f, N64);
  k_segmax<<<nb(EF * HID, 256), 256, 0, stream>>>(src, dst, h1, xqm);

  // xq = xqmax @ lin_W + lin_b
  gemm(xqm, HID, 0, LWT, HID, 0, linb, xq, HID, 1, 0, (int)(NN / 16), 1, 2, 1);

  k_att<<<nb(EF, 256), 256, 0, stream>>>(src, dst, xq, h1, attW, attb, sbuf);
  k_fill<<<nb(NN, 256), 256, 0, stream>>>(smax, -3.0e38f, NN);
  k_smax<<<nb(EF, 256), 256, 0, stream>>>(src, dst, sbuf, smax);
  k_fill<<<nb(NN, 256), 256, 0, stream>>>(ssum, 0.0f, NN);
  k_expsum<<<nb(EF, 256), 256, 0, stream>>>(src, dst, sbuf, smax, esc, ssum);
  k_norm<<<nb(EF, 256), 256, 0, stream>>>(src, dst, esc, ssum);

  k_fill<<<nb(N64, 256), 256, 0, stream>>>(xnew, 0.0f, N64);
  k_xnew<<<nb(EF * HID, 256), 256, 0, stream>>>(src, dst, h1, esc, xnew);

  // ---- fitness + top-k ----
  k_ledots<<<nb(NN, 256), 256, 0, stream>>>(xnew, le1W, le1b, le2W, le3W, le3b, aa, bb, cc);
  k_fill<<<nb(NN, 256), 256, 0, stream>>>(fit, 0.0f, NN);
  k_fitscatter<<<nb(EF, 256), 256, 0, stream>>>(src, dst, aa, fit);
  k_fitfin<<<nb(NN, 256), 256, 0, stream>>>(fit, deg, bb, cc);
  k_topk<<<dim3(BG), dim3(NPG), 0, stream>>>(fit, perm, kpos);
  k_xsel<<<nb((long)BG * KP * HID, 256), 256, 0, stream>>>(perm, xnew, fit, xsel);

  // ---- coarsening: dense A (aliases dead h/h1/xqm/xq), Ssel^T, batched WMMA ----
  k_fill<<<nb(4 * N64, 256), 256, 0, stream>>>(Aden, 0.0f, 4 * N64);
  k_fill<<<nb((long)BG * KP * NPG, 256), 256, 0, stream>>>(SselT, 0.0f, (long)BG * KP * NPG);
  k_Ascatter<<<nb(EF, 256), 256, 0, stream>>>(src, dst, Aden);
  k_Sselscatter<<<nb(EF, 256), 256, 0, stream>>>(src, dst, esc, kpos, SselT);

  // AS = A(256x256) @ Ssel(256x128); B from SselT (N-major), C stored as AS^T
  gemm(Aden, NPG, (long)NPG * NPG,
       SselT, NPG, (long)KP * NPG,
       nullptr, ASt, /*c_sM=*/1, /*c_sN=*/NPG, (long)KP * NPG,
       NPG / 16, KP / 64, NPG / 32, BG);
  // A2 = Ssel^T(128x256) @ AS(256x128); A from SselT (row-major now),
  // B from ASt (N-major); C row-major [i][k']
  gemm(SselT, NPG, (long)KP * NPG,
       ASt, NPG, (long)KP * NPG,
       nullptr, A2, KP, 1, (long)KP * KP,
       KP / 16, KP / 64, NPG / 32, BG);

  // ---- conv2 ----
  k_d2<<<nb((long)BG * KP, 256), 256, 0, stream>>>(A2, d2);
  k_Mt<<<nb((long)BG * KP * KP, 256), 256, 0, stream>>>(A2, d2, Mt);

  // h2 = xsel @ W2, batched over graphs, stored transposed h2t[g][f][m]
  gemm(xsel, HID, (long)KP * HID,
       W2T, HID, 0,
       nullptr, h2t, /*c_sM=*/1, /*c_sN=*/KP, (long)KP * HID,
       KP / 16, 1, 2, BG);
  // out2 = Mt(128x128) @ h2(128x64); B from h2t (N-major); C row-major [d][f]
  gemm(Mt, KP, (long)KP * KP,
       h2t, KP, (long)KP * HID,
       nullptr, out2, HID, 1, (long)KP * HID,
       KP / 16, 1, KP / 32, BG);

  k_reduce<<<nb((long)BG * HID, 256), 256, 0, stream>>>(out2, b2, out);
}
